// CausalSelfAttention_10196252361144
// MI455X (gfx1250) — compile-verified, matrix operand order checked by reading, not run on hardware
//
#include <hip/hip_runtime.h>
#include <hip/hip_bf16.h>
#include <cstdint>

// ---------------- problem constants ----------------
static constexpr int S_  = 2048;
static constexpr int D_  = 2048;
static constexpr int H_  = 16;
static constexpr int HKV_ = 4;
static constexpr int HD_ = 128;
static constexpr float SCALE_ = 0.08838834764831845f; // 1/sqrt(128)

// ---------------- types ----------------
typedef __attribute__((ext_vector_type(16))) __bf16 v16bf;
typedef __attribute__((ext_vector_type(8)))  __bf16 v8bf;
typedef __attribute__((ext_vector_type(8)))  float  v8f;

static __device__ __forceinline__ unsigned short f2bf(float f) {
    union { float f; unsigned u; } c; c.f = f;
    unsigned u = c.u + 0x7FFFu + ((c.u >> 16) & 1u);   // round-to-nearest-even
    return (unsigned short)(u >> 16);
}

static __device__ __forceinline__ v16bf frag_from(const __bf16* p0, const __bf16* p1) {
    union { v16bf v; v8bf h[2]; } u;
    u.h[0] = *(const v8bf*)p0;
    u.h[1] = *(const v8bf*)p1;
    return u.v;
}

static __device__ __forceinline__ v8f wmma_bf16(v16bf a, v16bf b, v8f c) {
    return __builtin_amdgcn_wmma_f32_16x16x32_bf16(false, a, false, b, (short)0, c,
                                                   false, false);
}

// ---------------- fp32 -> bf16 conversion (vectorized x4) ----------------
__global__ void k_cvt_bf16(const float4* __restrict__ in,
                           uint2* __restrict__ out, int n4) {
    int i = blockIdx.x * blockDim.x + threadIdx.x;
    if (i < n4) {
        float4 v = in[i];
        uint2 o;
        o.x = ((unsigned)f2bf(v.y) << 16) | f2bf(v.x);
        o.y = ((unsigned)f2bf(v.w) << 16) | f2bf(v.z);
        out[i] = o;
    }
}

// ---------------- WMMA GEMM: C[M,N] = A[M,K] * B[N,K]^T ----------------
// 256 threads = 8 waves laid out 2 (M) x 4 (N); wave tile 64x64 (4x4 WMMA tiles).
// Block tile 128 x 256.  grid: (N/256, M/128).  N, K compile-time constants so
// all fragment loads / C stores are base-pointer + 24-bit immediate offsets.
template <int N, int K>
__global__ __launch_bounds__(256) void k_gemm(
    const unsigned short* __restrict__ A, const unsigned short* __restrict__ B,
    float* __restrict__ C) {
    const int lane  = threadIdx.x & 31;
    const int wave  = threadIdx.x >> 5;
    const int laneM = lane & 15;
    const int hi    = lane >> 4;
    const int wm    = wave >> 2;      // 0..1
    const int wn    = wave & 3;       // 0..3
    const int rowBase = blockIdx.y * 128 + wm * 64;
    const int colBase = blockIdx.x * 256 + wn * 64;

    // base pointers include laneM row and hi*8 K-offset; all else is immediate
    const __bf16* a0 = (const __bf16*)A + (size_t)(rowBase + laneM) * K + hi * 8;
    const __bf16* b0 = (const __bf16*)B + (size_t)(colBase + laneM) * K + hi * 8;

    v8f acc[4][4];
#pragma unroll
    for (int i = 0; i < 4; ++i)
#pragma unroll
        for (int j = 0; j < 4; ++j) acc[i][j] = (v8f)0.f;

    for (int k0 = 0; k0 < K; k0 += 32) {
        v16bf af[4], bfr[4];
#pragma unroll
        for (int i = 0; i < 4; ++i)
            af[i] = frag_from(a0 + (size_t)i * 16 * K, a0 + (size_t)i * 16 * K + 16);
#pragma unroll
        for (int j = 0; j < 4; ++j)
            bfr[j] = frag_from(b0 + (size_t)j * 16 * K, b0 + (size_t)j * 16 * K + 16);
#pragma unroll
        for (int i = 0; i < 4; ++i)
#pragma unroll
            for (int j = 0; j < 4; ++j)
                acc[i][j] = wmma_bf16(af[i], bfr[j], acc[i][j]);
        a0 += 32;
        b0 += 32;
    }

    // epilogue: single base pointer, constant offsets
    float* cp = C + (size_t)(rowBase + 8 * hi) * N + colBase + laneM;
#pragma unroll
    for (int i = 0; i < 4; ++i)
#pragma unroll
        for (int j = 0; j < 4; ++j)
#pragma unroll
            for (int r = 0; r < 8; ++r)
                cp[(size_t)(i * 16 + r) * N + j * 16] = acc[i][j][r];
}

// ---------------- RMSNorm + RoPE (+gain), fp32 -> per-head bf16 ----------------
// in : [s][nheads*128] fp32; out: [h][s][128] bf16. grid(S, nheads), block 32.
__global__ __launch_bounds__(32) void k_rms_rope(
    const float* __restrict__ in, unsigned short* __restrict__ out,
    const float* __restrict__ gain, int nheads) {
    const int s = blockIdx.x, h = blockIdx.y, lane = threadIdx.x;
    const float* p = in + ((size_t)s * nheads + h) * HD_ + lane * 4;
    float4 x = *(const float4*)p;
    float ss = x.x * x.x + x.y * x.y + x.z * x.z + x.w * x.w;
#pragma unroll
    for (int m = 16; m >= 1; m >>= 1) ss += __shfl_xor(ss, m);
    float r = rsqrtf(ss * (1.f / 128.f) + 1.1920928955078125e-7f);
    float xn[4] = {x.x * r, x.y * r, x.z * r, x.w * r};
    float xp[4];
#pragma unroll
    for (int j = 0; j < 4; ++j) xp[j] = __shfl_xor(xn[j], 16);
    const float g = gain ? gain[h] : 1.f;
    unsigned short o[4];
#pragma unroll
    for (int j = 0; j < 4; ++j) {
        int d  = lane * 4 + j;
        int jj = (d < 64) ? d : (d - 64);
        float ang = (float)s * __powf(10000.f, -(float)jj * (1.f / 64.f));
        float c = __cosf(ang), si = __sinf(ang);
        float y = (d < 64) ? (xn[j] * c + xp[j] * si)
                           : (-xp[j] * si + xn[j] * c);
        o[j] = f2bf(y * g);
    }
    unsigned short* q = out + ((size_t)h * S_ + s) * HD_ + lane * 4;
    uint2 packed;
    packed.x = ((unsigned)o[1] << 16) | o[0];
    packed.y = ((unsigned)o[3] << 16) | o[2];
    *(uint2*)q = packed;
}

// ---------------- V transpose: [s][hk*128+d] fp32 -> [hk][d][s] bf16 ----------------
__global__ void k_vtrans(const float* __restrict__ in,
                         unsigned short* __restrict__ out) {
    int idx = blockIdx.x * blockDim.x + threadIdx.x;   // hk*HD*S total
    int s = idx & (S_ - 1);
    int d = (idx >> 11) & (HD_ - 1);
    int hk = idx >> 18;
    out[idx] = f2bf(in[(size_t)s * (HKV_ * HD_) + hk * HD_ + d]);
}

// ---------------- flash attention: one wave per (16-query block, head) ----------------
// q: [h][s][128] bf16, k: [hk][s][128] bf16, vt: [hk][128][s] bf16
// y: [s][H*128] fp32.  grid(S/16, H), block 32.
__global__ __launch_bounds__(32) void k_attn(
    const unsigned short* __restrict__ qb, const unsigned short* __restrict__ kb,
    const unsigned short* __restrict__ vt, float* __restrict__ y) {
    const int lane  = threadIdx.x;
    const int laneM = lane & 15;
    const int hi    = lane >> 4;
    const int q0    = blockIdx.x * 16;
    const int h     = blockIdx.y;
    const int hk    = h >> 2;   // G = 4

    const __bf16* qbase = (const __bf16*)qb + ((size_t)h * S_ + q0 + laneM) * HD_ + hi * 8;
    // kp: advances 32 rows (32*128 elems) per kv step; tile/chunk offsets immediate
    const __bf16* kp = (const __bf16*)kb + (size_t)hk * S_ * HD_ + (size_t)laneM * HD_ + hi * 8;
    // vp: advances 32 elems per kv step; hd-tile offsets immediate (t*16*S)
    const __bf16* vp = (const __bf16*)vt + (size_t)hk * HD_ * S_ + (size_t)laneM * S_ + hi * 8;

    v16bf qf[4];
#pragma unroll
    for (int c = 0; c < 4; ++c) qf[c] = frag_from(qbase + c * 32, qbase + c * 32 + 16);

    v8f acc[8];
#pragma unroll
    for (int t = 0; t < 8; ++t) acc[t] = (v8f)0.f;
    float mrow[8], lrow[8];
#pragma unroll
    for (int r = 0; r < 8; ++r) { mrow[r] = -__builtin_inff(); lrow[r] = 0.f; }

    __shared__ __align__(16) unsigned short Pl[16 * 32];

    const int nk = q0 + 16;
    for (int k0 = 0; k0 < nk; k0 += 32) {
        // prefetch next kv block (K rows and V columns) into cache
        __builtin_prefetch(kp + 32 * HD_, 0, 3);
        __builtin_prefetch(vp + 32, 0, 3);
        // ---- scores: two 16x16 tiles over 32 kv positions ----
        v8f sc[2];
#pragma unroll
        for (int n = 0; n < 2; ++n) {
            sc[n] = (v8f)0.f;
#pragma unroll
            for (int c = 0; c < 4; ++c)
                sc[n] = wmma_bf16(qf[c],
                                  frag_from(kp + n * 16 * HD_ + c * 32,
                                            kp + n * 16 * HD_ + c * 32 + 16),
                                  sc[n]);
        }
        const bool needMask = (k0 + 31 > q0);
#pragma unroll
        for (int n = 0; n < 2; ++n)
#pragma unroll
            for (int r = 0; r < 8; ++r) {
                float v = sc[n][r] * SCALE_;
                if (needMask) {
                    int kv = k0 + n * 16 + laneM;
                    int qi = q0 + r + 8 * hi;
                    if (kv > qi) v = -__builtin_inff();
                }
                sc[n][r] = v;
            }
        // ---- online softmax per row ----
#pragma unroll
        for (int r = 0; r < 8; ++r) {
            float mc = fmaxf(sc[0][r], sc[1][r]);
#pragma unroll
            for (int m = 8; m >= 1; m >>= 1) mc = fmaxf(mc, __shfl_xor(mc, m));
            float mn   = fmaxf(mrow[r], mc);
            float corr = __expf(mrow[r] - mn);
            float p0   = __expf(sc[0][r] - mn);
            float p1   = __expf(sc[1][r] - mn);
            float rs   = p0 + p1;
#pragma unroll
            for (int m = 8; m >= 1; m >>= 1) rs += __shfl_xor(rs, m);
            lrow[r] = lrow[r] * corr + rs;
            mrow[r] = mn;
#pragma unroll
            for (int t = 0; t < 8; ++t) acc[t][r] *= corr;
            int row = r + 8 * hi;
            Pl[row * 32 + laneM]      = f2bf(p0);
            Pl[row * 32 + 16 + laneM] = f2bf(p1);
        }
        // ---- transpose P via LDS into A-fragment layout (in-order per wave) ----
        const __bf16* pl = (const __bf16*)Pl + laneM * 32 + hi * 8;
        v16bf pf = frag_from(pl, pl + 16);
        // ---- P @ V ----
#pragma unroll
        for (int t = 0; t < 8; ++t)
            acc[t] = wmma_bf16(pf,
                               frag_from(vp + (size_t)t * 16 * S_,
                                         vp + (size_t)t * 16 * S_ + 16),
                               acc[t]);
        kp += 32 * HD_;
        vp += 32;
    }

    // ---- epilogue: y[s][h*128 + d] = acc / l ----
    float* yp = y + (size_t)(q0 + 8 * hi) * (H_ * HD_) + h * HD_ + laneM;
#pragma unroll
    for (int r = 0; r < 8; ++r) {
        float inv = 1.f / lrow[r];
#pragma unroll
        for (int t = 0; t < 8; ++t)
            yp[(size_t)r * (H_ * HD_) + t * 16] = acc[t][r] * inv;
    }
}

// ---------------- launcher ----------------
static inline size_t aln(size_t x) { return (x + 255) & ~(size_t)255; }

extern "C" void kernel_launch(void* const* d_in, const int* in_sizes, int n_in,
                              void* d_out, int out_size, void* d_ws, size_t ws_size,
                              hipStream_t stream) {
    (void)in_sizes; (void)n_in; (void)out_size; (void)ws_size;
    const float* x     = (const float*)d_in[0];
    const float* Wq    = (const float*)d_in[1];
    const float* Wk    = (const float*)d_in[2];
    const float* Wv    = (const float*)d_in[3];
    const float* Wp    = (const float*)d_in[4];
    const float* qgain = (const float*)d_in[5];
    float* out = (float*)d_out;

    // workspace carve
    uint8_t* w = (uint8_t*)d_ws;
    size_t off = 0;
    auto carve = [&](size_t bytes) { void* p = w + off; off += aln(bytes); return p; };
    unsigned short* xb  = (unsigned short*)carve((size_t)S_ * D_ * 2);
    unsigned short* wqb = (unsigned short*)carve((size_t)D_ * D_ * 2);
    unsigned short* wkb = (unsigned short*)carve((size_t)(HKV_ * HD_) * D_ * 2);
    unsigned short* wvb = (unsigned short*)carve((size_t)(HKV_ * HD_) * D_ * 2);
    unsigned short* wpb = (unsigned short*)carve((size_t)D_ * D_ * 2);
    float* Qf = (float*)carve((size_t)S_ * D_ * 4);
    float* Kf = (float*)carve((size_t)S_ * HKV_ * HD_ * 4);
    float* Vf = (float*)carve((size_t)S_ * HKV_ * HD_ * 4);
    unsigned short* qhb = (unsigned short*)carve((size_t)H_ * S_ * HD_ * 2);
    unsigned short* khb = (unsigned short*)carve((size_t)HKV_ * S_ * HD_ * 2);
    unsigned short* vtb = (unsigned short*)carve((size_t)HKV_ * HD_ * S_ * 2);
    float* yf = (float*)carve((size_t)S_ * D_ * 4);
    unsigned short* yb = (unsigned short*)carve((size_t)S_ * D_ * 2);

    const int TPB = 256;
    auto cvt = [&](const float* src, unsigned short* dst, int n) {
        int n4 = n / 4;
        k_cvt_bf16<<<(n4 + TPB - 1) / TPB, TPB, 0, stream>>>(
            (const float4*)src, (uint2*)dst, n4);
    };
    // 1) fp32 -> bf16 conversions
    cvt(x, xb, S_ * D_);
    cvt(Wq, wqb, D_ * D_);
    cvt(Wk, wkb, HKV_ * HD_ * D_);
    cvt(Wv, wvb, HKV_ * HD_ * D_);
    cvt(Wp, wpb, D_ * D_);

    // 2) QKV projections (WMMA GEMM, B given as [N][K] row-major = W)
    k_gemm<D_, D_><<<dim3(D_ / 256, S_ / 128), 256, 0, stream>>>(xb, wqb, Qf);
    k_gemm<HKV_ * HD_, D_><<<dim3((HKV_ * HD_) / 256, S_ / 128), 256, 0, stream>>>(xb, wkb, Kf);
    k_gemm<HKV_ * HD_, D_><<<dim3((HKV_ * HD_) / 256, S_ / 128), 256, 0, stream>>>(xb, wvb, Vf);

    // 3) RMSNorm + RoPE (+gain for Q), to per-head bf16
    k_rms_rope<<<dim3(S_, H_), 32, 0, stream>>>(Qf, qhb, qgain, H_);
    k_rms_rope<<<dim3(S_, HKV_), 32, 0, stream>>>(Kf, khb, nullptr, HKV_);

    // 4) V transpose to [hk][hd][s] bf16
    k_vtrans<<<(HKV_ * HD_ * S_) / TPB, TPB, 0, stream>>>(Vf, vtb);

    // 5) flash attention
    k_attn<<<dim3(S_ / 16, H_), 32, 0, stream>>>(qhb, khb, vtb, yf);

    // 6) output projection
    cvt(yf, yb, S_ * D_);
    k_gemm<D_, D_><<<dim3(D_ / 256, S_ / 128), 256, 0, stream>>>(yb, wpb, out);
}
